// SlidingWindowAttention_5102421147672
// MI455X (gfx1250) — compile-verified, matrix operand order checked by reading, not run on hardware
//
#include <hip/hip_runtime.h>
#include <hip/hip_bf16.h>

typedef __attribute__((ext_vector_type(16))) _Float16 v16h;
typedef __attribute__((ext_vector_type(8)))  _Float16 v8h;
typedef __attribute__((ext_vector_type(8)))  float    v8f;

namespace cfg {
constexpr int B = 2, T = 2048, D = 1024, H = 16, KD = 64, V = 64, WIN = 256;
constexpr int BANK = 64, CUE = 50250;
constexpr float EPS = 1e-6f, SCALE = 0.125f, LOCAL_SCALE = 0.3f;
constexpr int MT = B * T;  // 4096 rows
}

// ---------------------------------------------------------------------------
// WMMA helpers. CDNA5 16x16x32 f16 layout (CDNA5 ISA):
//   A (16x32, row-major in K): lane l holds row (l&15); VGPR0-3 = K kh..kh+7,
//   VGPR4-7 = K 16+kh..16+kh+7, kh = (l>>4)*8  -> two contiguous 16B loads.
//   Same pattern serves the B operand of A*B^T when B is row-major in K,
//   or any pre-transposed KxN matrix stored as [N][K].
//   C/D: VGPR j -> row j + (l>>4)*8, col (l&15).
// ---------------------------------------------------------------------------
__device__ __forceinline__ v16h load_frag(const _Float16* base, int ld) {
  const int lane = threadIdx.x & 31;
  const _Float16* p = base + (lane & 15) * ld + ((lane >> 4) << 3);
  v8h lo = *(const v8h*)p;
  v8h hi = *(const v8h*)(p + 16);
  v16h f;
#pragma unroll
  for (int i = 0; i < 8; ++i) { f[i] = lo[i]; f[i + 8] = hi[i]; }
  return f;
}

__device__ __forceinline__ v8f wmma16(v16h a, v16h b, v8f c) {
  return __builtin_amdgcn_wmma_f32_16x16x32_f16(false, a, false, b,
                                                (short)0, c, false, false);
}

// Async cache->LDS copy (CDNA5 GLOBAL_LOAD_ASYNC_TO_LDS_B128, ASYNCcnt).
// vdst = low 32 bits of flat LDS address (= LDS byte offset per aperture
// mapping); vaddr = 64-bit global address (GV mode).
__device__ __forceinline__ void async_copy_b128(void* ldsPtr, const void* gptr) {
  const unsigned loff = (unsigned)(unsigned long long)ldsPtr;
  asm volatile("global_load_async_to_lds_b128 %0, %1, off"
               :: "v"(loff), "v"((unsigned long long)gptr) : "memory");
}
__device__ __forceinline__ void async_wait0() {
  asm volatile("s_wait_asynccnt 0x0" ::: "memory");
}

// ---------------------------------------------------------------------------
// 1) RMS norm (f32 in) -> x_norm (f16 out)
// ---------------------------------------------------------------------------
__global__ __launch_bounds__(256) void rmsnorm_kernel(
    const float* __restrict__ x, const float* __restrict__ nw,
    _Float16* __restrict__ xn) {
  using namespace cfg;
  const int row = blockIdx.x;
  const float* xr = x + (size_t)row * D;
  float ss = 0.f;
  for (int d = threadIdx.x; d < D; d += 256) { float v = xr[d]; ss += v * v; }
#pragma unroll
  for (int off = 16; off; off >>= 1) ss += __shfl_xor(ss, off, 32);
  __shared__ float red[8];
  if ((threadIdx.x & 31) == 0) red[threadIdx.x >> 5] = ss;
  __syncthreads();
  float tot = 0.f;
#pragma unroll
  for (int i = 0; i < 8; ++i) tot += red[i];
  const float rinv = rsqrtf(tot / (float)D + EPS);
  for (int d = threadIdx.x; d < D; d += 256)
    xn[(size_t)row * D + d] = (_Float16)(xr[d] * rinv * nw[d]);
}

// ---------------------------------------------------------------------------
// 2) f32 -> f16 conversion (weights)
// ---------------------------------------------------------------------------
__global__ __launch_bounds__(256) void cvt_f32_f16(
    const float* __restrict__ a, _Float16* __restrict__ o, int n) {
  int i = blockIdx.x * 256 + threadIdx.x;
  if (i < n) o[i] = (_Float16)a[i];
}

// ---------------------------------------------------------------------------
// 3) Generic NT GEMM: C[M x 1024](f16) = A[M x 1024](f16) * W[1024 x 1024]^T
//    block = 8 waves, tile 256x64 (32 rows/wave). W tile (64x32) staged in
//    double-buffered LDS via GLOBAL_LOAD_ASYNC_TO_LDS_B128: the copy for
//    k-step i+1 overlaps the 8 WMMAs of k-step i; s_wait_asynccnt lands
//    after the compute instead of right after issue.
// ---------------------------------------------------------------------------
__global__ __launch_bounds__(256) void gemm_nt_f16(
    const _Float16* __restrict__ A, const _Float16* __restrict__ W,
    _Float16* __restrict__ C) {
  constexpr int Kd = 1024, N = 1024;
  __shared__ __align__(16) _Float16 wt[2][64 * 32];
  const int wave = threadIdx.x >> 5, lane = threadIdx.x & 31;
  const int row0 = blockIdx.x * 256 + wave * 32;
  const int col0 = blockIdx.y * 64;
  v8f acc[2][4] = {};
  const int tr = threadIdx.x >> 2, tc = (threadIdx.x & 3) * 8;
  void* ldst[2] = { &wt[0][tr * 32 + tc], &wt[1][tr * 32 + tc] };
  const _Float16* wrow = &W[(size_t)(col0 + tr) * Kd + tc];

  async_copy_b128(ldst[0], wrow);           // prologue: stage k0 = 0
  async_wait0();
  __syncthreads();

  for (int k0 = 0; k0 < Kd; k0 += 32) {
    const int cur = (k0 >> 5) & 1;
    if (k0 + 32 < Kd)                        // stage next tile into alt buffer
      async_copy_b128(ldst[cur ^ 1], wrow + k0 + 32);
    const v16h a0 = load_frag(A + (size_t)row0 * Kd + k0, Kd);
    const v16h a1 = load_frag(A + (size_t)(row0 + 16) * Kd + k0, Kd);
#pragma unroll
    for (int n = 0; n < 4; ++n) {
      const v16h b = load_frag(&wt[cur][n * 16 * 32], 32);
      acc[0][n] = wmma16(a0, b, acc[0][n]);
      acc[1][n] = wmma16(a1, b, acc[1][n]);
    }
    async_wait0();                           // next tile resident
    __syncthreads();                         // all waves done reading cur
  }
  const int r = (lane >> 4) * 8, c = lane & 15;
#pragma unroll
  for (int hf = 0; hf < 2; ++hf)
#pragma unroll
    for (int n = 0; n < 4; ++n)
#pragma unroll
      for (int j = 0; j < 8; ++j)
        C[(size_t)(row0 + hf * 16 + r + j) * N + col0 + n * 16 + c] =
            (_Float16)acc[hf][n][j];
}

// ---------------------------------------------------------------------------
// 4) RoPE in place on a (B,T,H,64) f16 buffer; one thread per (i, i+32) pair.
// ---------------------------------------------------------------------------
__global__ __launch_bounds__(256) void rope_kernel(_Float16* __restrict__ u, int total) {
  using namespace cfg;
  int idx = blockIdx.x * 256 + threadIdx.x;
  if (idx >= total) return;
  const int j = idx & 31, h = (idx >> 5) & 15, t = (idx >> 9) & (T - 1), b = idx >> 20;
  const float inv = __powf(10000.f, -(float)(2 * j) / 64.f);
  float c, s;
  __sincosf((float)t * inv, &s, &c);  // s=sin, c=cos
  const size_t base = (size_t)(b * T + t) * D + h * 64;
  const float u1 = (float)u[base + j], u2 = (float)u[base + 32 + j];
  u[base + j]      = (_Float16)(u1 * c - u2 * s);
  u[base + 32 + j] = (_Float16)(u2 * c + u1 * s);
}

// ---------------------------------------------------------------------------
// 5) Transpose v: (B,T,H,V) -> vt (B,H,V,T)   (for P*V B-operand loads)
// ---------------------------------------------------------------------------
__global__ __launch_bounds__(256) void transposeV_kernel(
    const _Float16* __restrict__ v, _Float16* __restrict__ vt, int total) {
  using namespace cfg;
  int idx = blockIdx.x * 256 + threadIdx.x;
  if (idx >= total) return;
  const int t = idx & (T - 1), d = (idx >> 11) & 63, bh = idx >> 17;
  vt[idx] = v[(size_t)((bh >> 4) * T + t) * D + (bh & 15) * 64 + d];
}

// 5b) Transpose gdn_state (B,H,K,V) f32 -> (B,H,V,K) f16
__global__ __launch_bounds__(256) void transposeState_kernel(
    const float* __restrict__ st, _Float16* __restrict__ stT, int total) {
  int idx = blockIdx.x * 256 + threadIdx.x;
  if (idx >= total) return;
  const int k = idx & 63, vv = (idx >> 6) & 63, bh = idx >> 12;
  stT[idx] = (_Float16)st[((size_t)bh * 64 + k) * 64 + vv];
}

// ---------------------------------------------------------------------------
// 6) Cue gather: overwrite q_g rows where input id hits the key bank
// ---------------------------------------------------------------------------
__global__ __launch_bounds__(256) void gather_kernel(
    const int* __restrict__ ids, const float* __restrict__ bank,
    _Float16* __restrict__ qg, int total) {
  using namespace cfg;
  int idx = blockIdx.x * 256 + threadIdx.x;
  if (idx >= total) return;
  const int k = idx & 63, h = (idx >> 6) & 15, t = (idx >> 10) & (T - 1), b = idx >> 21;
  const int id = ids[b * T + t] - CUE;
  if (id < 0 || id >= BANK) return;
  qg[(size_t)(b * T + t) * D + h * 64 + k] = (_Float16)bank[((h * BANK + id) * 64) + k];
}

// ---------------------------------------------------------------------------
// 7) Sliding-window flash attention. One wave per 16-row query tile.
//    grid = (T/128, B*H), 8 waves/block. Keys in 32-chunks, <=10 per tile.
// ---------------------------------------------------------------------------
__global__ __launch_bounds__(256) void attn_kernel(
    const _Float16* __restrict__ q, const _Float16* __restrict__ k,
    const _Float16* __restrict__ vt, _Float16* __restrict__ localO) {
  using namespace cfg;
  __shared__ __align__(16) _Float16 pt[8][16 * 32];
  const int wave = threadIdx.x >> 5, lane = threadIdx.x & 31;
  const int bh = blockIdx.y, b = bh >> 4, h = bh & 15;
  const int qs = (blockIdx.x * 8 + wave) * 16;
  const int colb = lane & 15, rbase = (lane >> 4) << 3;

  const _Float16* qbase = q + (size_t)(b * T + qs) * D + h * 64;
  const v16h qf0 = load_frag(qbase, D);
  const v16h qf1 = load_frag(qbase + 32, D);

  v8f acc[4] = {};
  float m[8], l[8];
#pragma unroll
  for (int j = 0; j < 8; ++j) { m[j] = -1e30f; l[j] = 0.f; }

  const int smin = qs - WIN > 0 ? qs - WIN : 0;
  for (int c = smin & ~31; c <= qs + 15; c += 32) {
    const _Float16* kb = k + (size_t)(b * T + c) * D + h * 64;
    v8f s[2] = {};
#pragma unroll
    for (int hh = 0; hh < 2; ++hh) {
      s[hh] = wmma16(qf0, load_frag(kb + hh * 16 * D, D), s[hh]);
      s[hh] = wmma16(qf1, load_frag(kb + hh * 16 * D + 32, D), s[hh]);
    }
#pragma unroll
    for (int j = 0; j < 8; ++j) {
      const int t = qs + rbase + j;
      float s0 = s[0][j] * SCALE, s1 = s[1][j] * SCALE;
      const int i0 = c + colb, i1 = c + 16 + colb;
      if (i0 > t || i0 < t - WIN) s0 = -1e30f;
      if (i1 > t || i1 < t - WIN) s1 = -1e30f;
      float mr = s0 > s1 ? s0 : s1;
#pragma unroll
      for (int off = 1; off < 16; off <<= 1) {
        const float o = __shfl_xor(mr, off, 16);
        mr = mr > o ? mr : o;
      }
      const float mnew = m[j] > mr ? m[j] : mr;
      const float alpha = __expf(m[j] - mnew);
      const float p0 = __expf(s0 - mnew), p1 = __expf(s1 - mnew);
      pt[wave][(rbase + j) * 32 + colb]      = (_Float16)p0;
      pt[wave][(rbase + j) * 32 + 16 + colb] = (_Float16)p1;
      float rs = p0 + p1;
#pragma unroll
      for (int off = 1; off < 16; off <<= 1) rs += __shfl_xor(rs, off, 16);
      l[j] = l[j] * alpha + rs;
      m[j] = mnew;
#pragma unroll
      for (int n = 0; n < 4; ++n) acc[n][j] *= alpha;
    }
    const v16h pf = load_frag(&pt[wave][0], 32);
#pragma unroll
    for (int n = 0; n < 4; ++n)
      acc[n] = wmma16(pf, load_frag(vt + ((size_t)bh * 64 + n * 16) * T + c, T), acc[n]);
  }
#pragma unroll
  for (int n = 0; n < 4; ++n)
#pragma unroll
    for (int j = 0; j < 8; ++j)
      localO[(size_t)(b * T + qs + rbase + j) * D + h * 64 + n * 16 + colb] =
          (_Float16)(acc[n][j] / l[j]);
}

// ---------------------------------------------------------------------------
// 8) gdn retrieval: retrieved[b,t,h,:] = q_g[b,t,h,:] @ state[b,h] (64x64)
//    stT is (B,H,V,K) f16 so B-operand loads are contiguous-B128.
// ---------------------------------------------------------------------------
__global__ __launch_bounds__(256) void gdn_kernel(
    const _Float16* __restrict__ qg, const _Float16* __restrict__ stT,
    _Float16* __restrict__ retr) {
  using namespace cfg;
  const int wave = threadIdx.x >> 5, lane = threadIdx.x & 31;
  const int bh = blockIdx.y, b = bh >> 4, h = bh & 15;
  const int t0 = (blockIdx.x * 8 + wave) * 16;
  const _Float16* ab = qg + (size_t)(b * T + t0) * D + h * 64;
  const v16h a0 = load_frag(ab, D), a1 = load_frag(ab + 32, D);
  const _Float16* sb = stT + (size_t)bh * 64 * 64;
  v8f acc[4] = {};
#pragma unroll
  for (int n = 0; n < 4; ++n) {
    acc[n] = wmma16(a0, load_frag(sb + n * 16 * 64, 64), acc[n]);
    acc[n] = wmma16(a1, load_frag(sb + n * 16 * 64 + 32, 64), acc[n]);
  }
  const int r = (lane >> 4) * 8, c = lane & 15;
#pragma unroll
  for (int n = 0; n < 4; ++n)
#pragma unroll
    for (int j = 0; j < 8; ++j)
      retr[(size_t)(b * T + t0 + r + j) * D + h * 64 + n * 16 + c] =
          (_Float16)acc[n][j];
}

// ---------------------------------------------------------------------------
// 9) Gate: gmean[row] = mean_h sigmoid(x_norm . w_gate[h] + b_gate[h])
// ---------------------------------------------------------------------------
__global__ __launch_bounds__(256) void gate_kernel(
    const _Float16* __restrict__ xn, const float* __restrict__ wg,
    const float* __restrict__ bg, float* __restrict__ gmean) {
  using namespace cfg;
  const int row = blockIdx.x;
  const int wave = threadIdx.x >> 5, lane = threadIdx.x & 31;
  __shared__ float acc8[8];
  float local = 0.f;
#pragma unroll
  for (int h = wave; h < H; h += 8) {
    float s = 0.f;
    const float* w = wg + (size_t)h * D;
    for (int d = lane; d < D; d += 32) s += (float)xn[(size_t)row * D + d] * w[d];
#pragma unroll
    for (int off = 16; off; off >>= 1) s += __shfl_xor(s, off, 32);
    if (lane == 0) local += 1.f / (1.f + __expf(-(s + bg[h])));
  }
  if (lane == 0) acc8[wave] = local;
  __syncthreads();
  if (threadIdx.x == 0) {
    float t = 0.f;
#pragma unroll
    for (int i = 0; i < 8; ++i) t += acc8[i];
    gmean[row] = t / (float)H;
  }
}

// ---------------------------------------------------------------------------
// 10) Fused epilogue: out = x + 0.3*(local@wo^T) + gmean[m]*(retr@w_ro^T)
//     Both W tiles staged with double-buffered async cache->LDS copies.
// ---------------------------------------------------------------------------
__global__ __launch_bounds__(256) void epilogue_gemm(
    const _Float16* __restrict__ A1, const _Float16* __restrict__ W1,
    const _Float16* __restrict__ A2, const _Float16* __restrict__ W2,
    const float* __restrict__ x, const float* __restrict__ gmean,
    float* __restrict__ out) {
  using namespace cfg;
  constexpr int Kd = 1024, N = 1024;
  __shared__ __align__(16) _Float16 w1t[2][64 * 32];
  __shared__ __align__(16) _Float16 w2t[2][64 * 32];
  const int wave = threadIdx.x >> 5, lane = threadIdx.x & 31;
  const int row0 = blockIdx.x * 128 + wave * 16;
  const int col0 = blockIdx.y * 64;
  v8f a1[4] = {}, a2[4] = {};
  const int tr = threadIdx.x >> 2, tc = (threadIdx.x & 3) * 8;
  void* l1[2] = { &w1t[0][tr * 32 + tc], &w1t[1][tr * 32 + tc] };
  void* l2[2] = { &w2t[0][tr * 32 + tc], &w2t[1][tr * 32 + tc] };
  const _Float16* w1row = &W1[(size_t)(col0 + tr) * Kd + tc];
  const _Float16* w2row = &W2[(size_t)(col0 + tr) * Kd + tc];

  async_copy_b128(l1[0], w1row);
  async_copy_b128(l2[0], w2row);
  async_wait0();
  __syncthreads();

  for (int k0 = 0; k0 < Kd; k0 += 32) {
    const int cur = (k0 >> 5) & 1;
    if (k0 + 32 < Kd) {
      async_copy_b128(l1[cur ^ 1], w1row + k0 + 32);
      async_copy_b128(l2[cur ^ 1], w2row + k0 + 32);
    }
    const v16h f1 = load_frag(A1 + (size_t)row0 * Kd + k0, Kd);
    const v16h f2 = load_frag(A2 + (size_t)row0 * Kd + k0, Kd);
#pragma unroll
    for (int n = 0; n < 4; ++n) {
      a1[n] = wmma16(f1, load_frag(&w1t[cur][n * 16 * 32], 32), a1[n]);
      a2[n] = wmma16(f2, load_frag(&w2t[cur][n * 16 * 32], 32), a2[n]);
    }
    async_wait0();
    __syncthreads();
  }
  const int r = (lane >> 4) * 8, c = lane & 15;
#pragma unroll
  for (int n = 0; n < 4; ++n)
#pragma unroll
    for (int j = 0; j < 8; ++j) {
      const int row = row0 + r + j;
      const size_t idx = (size_t)row * N + col0 + n * 16 + c;
      out[idx] = x[idx] + LOCAL_SCALE * a1[n][j] + gmean[row] * a2[n][j];
    }
}

// ---------------------------------------------------------------------------
extern "C" void kernel_launch(void* const* d_in, const int* in_sizes, int n_in,
                              void* d_out, int out_size, void* d_ws, size_t ws_size,
                              hipStream_t stream) {
  using namespace cfg;
  const float* x      = (const float*)d_in[0];
  const float* gdn    = (const float*)d_in[1];
  const int*   ids    = (const int*)d_in[2];
  const float* bank   = (const float*)d_in[3];
  const float* norm_w = (const float*)d_in[4];
  const float* wq     = (const float*)d_in[5];
  const float* wk     = (const float*)d_in[6];
  const float* wv     = (const float*)d_in[7];
  const float* wo     = (const float*)d_in[8];
  const float* wgq    = (const float*)d_in[9];
  const float* wro    = (const float*)d_in[10];
  const float* wgate  = (const float*)d_in[11];
  const float* bgate  = (const float*)d_in[12];
  float* out = (float*)d_out;

  const size_t ACT = (size_t)MT * D;        // 4M halfs per activation buffer
  const size_t WSZ = (size_t)D * D;         // 1M halfs per weight
  _Float16* xn   = (_Float16*)d_ws;
  _Float16* wq16 = xn + ACT;
  _Float16* wk16 = wq16 + WSZ;
  _Float16* wv16 = wk16 + WSZ;
  _Float16* wo16 = wv16 + WSZ;
  _Float16* wgq16 = wo16 + WSZ;
  _Float16* wro16 = wgq16 + WSZ;
  _Float16* q16  = wro16 + WSZ;
  _Float16* k16  = q16 + ACT;
  _Float16* v16  = k16 + ACT;
  _Float16* vt16 = v16 + ACT;
  _Float16* qg16 = vt16 + ACT;
  _Float16* st16 = qg16 + ACT;
  _Float16* re16 = st16 + (size_t)B * H * 64 * 64;
  _Float16* lo16 = re16 + ACT;
  float* gmean   = (float*)(lo16 + ACT);

  const dim3 blk(256);
  rmsnorm_kernel<<<MT, blk, 0, stream>>>(x, norm_w, xn);

  const int wn = (int)WSZ;
  cvt_f32_f16<<<(wn + 255) / 256, blk, 0, stream>>>(wq,  wq16,  wn);
  cvt_f32_f16<<<(wn + 255) / 256, blk, 0, stream>>>(wk,  wk16,  wn);
  cvt_f32_f16<<<(wn + 255) / 256, blk, 0, stream>>>(wv,  wv16,  wn);
  cvt_f32_f16<<<(wn + 255) / 256, blk, 0, stream>>>(wo,  wo16,  wn);
  cvt_f32_f16<<<(wn + 255) / 256, blk, 0, stream>>>(wgq, wgq16, wn);
  cvt_f32_f16<<<(wn + 255) / 256, blk, 0, stream>>>(wro, wro16, wn);

  const dim3 ggrid(MT / 256, D / 64);  // (16,16)
  gemm_nt_f16<<<ggrid, blk, 0, stream>>>(xn, wq16,  q16);
  gemm_nt_f16<<<ggrid, blk, 0, stream>>>(xn, wk16,  k16);
  gemm_nt_f16<<<ggrid, blk, 0, stream>>>(xn, wv16,  v16);
  gemm_nt_f16<<<ggrid, blk, 0, stream>>>(xn, wgq16, qg16);

  const int pairs = B * T * H * 32;  // 2,097,152
  rope_kernel<<<(pairs + 255) / 256, blk, 0, stream>>>(q16, pairs);
  rope_kernel<<<(pairs + 255) / 256, blk, 0, stream>>>(k16, pairs);

  const int actN = (int)ACT;  // 4,194,304
  transposeV_kernel<<<(actN + 255) / 256, blk, 0, stream>>>(v16, vt16, actN);
  const int stN = B * H * 64 * 64;  // 131,072
  transposeState_kernel<<<(stN + 255) / 256, blk, 0, stream>>>(gdn, st16, stN);
  gather_kernel<<<(actN + 255) / 256, blk, 0, stream>>>(ids, bank, qg16, actN);

  const dim3 agrid(T / 128, B * H);  // (16,32)
  attn_kernel<<<agrid, blk, 0, stream>>>(q16, k16, vt16, lo16);
  gdn_kernel<<<agrid, blk, 0, stream>>>(qg16, st16, re16);
  gate_kernel<<<MT, blk, 0, stream>>>(xn, wgate, bgate, gmean);

  const dim3 egrid(MT / 128, D / 64);  // (32,16)
  epilogue_gemm<<<egrid, blk, 0, stream>>>(lo16, wo16, re16, wro16, x, gmean, out);
}
